// MultiHeadAttention_39436389712272
// MI455X (gfx1250) — compile-verified, matrix operand order checked by reading, not run on hardware
//
#include <hip/hip_runtime.h>
#include <hip/hip_bf16.h>

typedef __attribute__((ext_vector_type(16))) __bf16 v16bf;
typedef __attribute__((ext_vector_type(8)))  float  v8f;
typedef __attribute__((ext_vector_type(4)))  int    i4x;
typedef __attribute__((ext_vector_type(4)))  float  f4x;
typedef __attribute__((ext_vector_type(4)))  unsigned int u32x4;
typedef __attribute__((ext_vector_type(8)))  int    i32x8;
typedef __attribute__((ext_vector_type(4)))  int    i32x4;
typedef unsigned short u16;
typedef unsigned int   u32;
typedef unsigned long long u64;

#define DEV static __device__ __forceinline__

// Problem dims
#define SEQ 1024
#define BATCH 8
#define EMB 1024
#define NH 16
#define DH 64
#define NEGBIG -1e9f

// ---------------------------------------------------------------------------
// CDNA5 async / TDM feature probes (compile-safe: manual path is the default)
// ---------------------------------------------------------------------------
#if __has_builtin(__builtin_amdgcn_global_load_async_to_lds_b128)
#define HAS_ASYNC 1
// Builtin signature (from clang diagnostic): param0 = AS(1) pointer to
// gcc-vector v4i (printed as "__device__"), so use vector_size(16) int.
typedef int v4i_g __attribute__((vector_size(16)));
typedef __attribute__((address_space(1))) v4i_g* gptr_v4i;
typedef __attribute__((address_space(3))) v4i_g* lptr_v4i;
DEV void async_copy16(void* dst_lds, const void* src_global) {
  __builtin_amdgcn_global_load_async_to_lds_b128(
      (gptr_v4i)(u64)(size_t)src_global,
      (lptr_v4i)(u32)(size_t)dst_lds,   // flat LDS addr low 32 bits = offset
      0, 0);
}
DEV void wait_async0() {
#if __has_builtin(__builtin_amdgcn_s_wait_asynccnt)
  __builtin_amdgcn_s_wait_asynccnt(0);
#else
  asm volatile("s_wait_asynccnt 0x0" ::: "memory");
#endif
}
#endif

#if __has_builtin(__builtin_amdgcn_tensor_load_to_lds) && \
    __has_builtin(__builtin_amdgcn_s_wait_tensorcnt)
#define HAS_TDM 1
#if __has_include(<hip/amd_detail/amd_gfx1250_TDM.h>)
#define TDM_ARITY 6
#else
#define TDM_ARITY 5
#endif
#endif

DEV u16 f2bf(float f) {
  u32 u = __float_as_uint(f);
  u32 r = u + 0x7fffu + ((u >> 16) & 1u);
  return (u16)(r >> 16);
}

union FragU {
  i4x  i4[2];
  u16  u[16];
  v16bf v;
};

// Load a 16x32 bf16 WMMA fragment (A layout: lane=row, elems=K; B layout:
// lane=col(N), elems=K -- same per-lane addressing) from row-major [row][K]
// storage with `pitch` elements per row. Element e<8 -> K = kbase+8*hi+e,
// e>=8 -> K = kbase+16+8*hi+(e-8), matching the CDNA5 ISA 16-bit layout.
DEV v16bf load_frag(const u16* base, int pitch, int row, int kbase, int lane) {
  const int r = lane & 15, hi = lane >> 4;
  const u16* p = base + (size_t)(row + r) * pitch + kbase + hi * 8;
  FragU f;
  f.i4[0] = *(const i4x*)(p);
  f.i4[1] = *(const i4x*)(p + 16);
  return f.v;
}

DEV v8f wmma_bf(v16bf a, v16bf b, v8f c) {
  return __builtin_amdgcn_wmma_f32_16x16x32_bf16(false, a, false, b, (short)0, c,
                                                 false, false);
}

// ---------------------------------------------------------------------------
// f32 -> bf16 conversion kernel
// ---------------------------------------------------------------------------
__global__ __launch_bounds__(256) void cvt_bf16_kernel(const float* __restrict__ src,
                                                       u16* __restrict__ dst, int n) {
  int i = blockIdx.x * 256 + threadIdx.x;
  if (i < n) dst[i] = f2bf(src[i]);
}

// ---------------------------------------------------------------------------
// Shared 128x128 block-tile bf16 GEMM core: C[m][n] = sum_k A[m][k]*W[n][k]
// 8 waves in a 2x4 grid, each wave owns a 64x32 sub-tile (4x2 16x16 frags).
// ---------------------------------------------------------------------------
#define GP 40   // LDS pitch (32 + 8 pad) in u16 elements, keeps b128 alignment

DEV void gemm128(const u16* __restrict__ A, const u16* __restrict__ W,
                 int m0, int n0, u16* As, u16* Bs, v8f acc[4][2]) {
  const int tid = threadIdx.x, lane = tid & 31, w = tid >> 5;
  const int wy = w >> 2, wx = w & 3;
  const int srow = tid >> 1, skq = (tid & 1) * 16;   // staging: 2 threads/row

  for (int k0 = 0; k0 < EMB; k0 += 32) {
    __syncthreads();
    {
      const u16* pa = A + (size_t)(m0 + srow) * EMB + k0 + skq;
      const u16* pw = W + (size_t)(n0 + srow) * EMB + k0 + skq;
      u16* da = As + srow * GP + skq;
      u16* dw = Bs + srow * GP + skq;
#if defined(HAS_ASYNC)
      async_copy16(da, pa);
      async_copy16(da + 8, pa + 8);
      async_copy16(dw, pw);
      async_copy16(dw + 8, pw + 8);
#else
      *(i4x*)da       = *(const i4x*)pa;
      *(i4x*)(da + 8) = *(const i4x*)(pa + 8);
      *(i4x*)dw       = *(const i4x*)pw;
      *(i4x*)(dw + 8) = *(const i4x*)(pw + 8);
#endif
      __builtin_prefetch(pa + 32, 0, 1);
      __builtin_prefetch(pw + 32, 0, 1);
#if defined(HAS_ASYNC)
      wait_async0();
#endif
    }
    __syncthreads();

    v16bf af[4], bfg[2];
#pragma unroll
    for (int mt = 0; mt < 4; ++mt)
      af[mt] = load_frag(As, GP, wy * 64 + mt * 16, 0, lane);
#pragma unroll
    for (int nt = 0; nt < 2; ++nt)
      bfg[nt] = load_frag(Bs, GP, wx * 32 + nt * 16, 0, lane);
#pragma unroll
    for (int mt = 0; mt < 4; ++mt)
#pragma unroll
      for (int nt = 0; nt < 2; ++nt)
        acc[mt][nt] = wmma_bf(af[mt], bfg[nt], acc[mt][nt]);
  }
}

// ---------------------------------------------------------------------------
// QKV projection: Y = x @ qkv_w^T + b ; scatter to per-head q/k/v bf16 with
// mask and 1/sqrt(Dh) folded in.
// ---------------------------------------------------------------------------
__global__ __launch_bounds__(256) void gemm_qkv_kernel(
    const u16* __restrict__ xb, const u16* __restrict__ wqb,
    const float* __restrict__ qkvb, const int* __restrict__ mask,
    u16* __restrict__ qb, u16* __restrict__ kb, u16* __restrict__ vb) {
  __shared__ u16 As[128 * GP];
  __shared__ u16 Bs[128 * GP];
  const int m0 = blockIdx.y * 128, n0 = blockIdx.x * 128;
  const int tid = threadIdx.x, lane = tid & 31, w = tid >> 5;
  const int wy = w >> 2, wx = w & 3, r = lane & 15, hi = lane >> 4;

  v8f acc[4][2] = {};
  gemm128(xb, wqb, m0, n0, As, Bs, acc);

#pragma unroll
  for (int mt = 0; mt < 4; ++mt) {
#pragma unroll
    for (int nt = 0; nt < 2; ++nt) {
      const int n = n0 + wx * 32 + nt * 16 + r;
      const float bias = qkvb[n];
      const int sel = n >> 10;           // 0=q 1=k 2=v
      const int nn = n & (EMB - 1);
      const int head = nn >> 6, d = nn & 63;
#pragma unroll
      for (int e = 0; e < 8; ++e) {
        const int m = m0 + wy * 64 + mt * 16 + e + 8 * hi;
        const int s = m >> 3, bb = m & 7;               // m = s*B + b
        const float y = acc[mt][nt][e] + bias;
        const size_t oidx = (((size_t)(bb * NH + head)) * SEQ + s) * DH + d;
        if (sel == 0) {
          const float mv = (float)mask[bb * SEQ + s];
          qb[oidx] = f2bf(y * 0.125f * mv);             // 1/sqrt(64)
        } else if (sel == 1) {
          const float mv = (float)mask[bb * SEQ + s];
          kb[oidx] = f2bf(y * mv);
        } else {
          vb[oidx] = f2bf(y);
        }
      }
    }
  }
}

// ---------------------------------------------------------------------------
// Output projection: o = oh @ out_w^T + b -> d_out[(s*B+b)*E + n]
// ---------------------------------------------------------------------------
__global__ __launch_bounds__(256) void gemm_out_kernel(
    const u16* __restrict__ ohb, const u16* __restrict__ wob,
    const float* __restrict__ outb, float* __restrict__ o) {
  __shared__ u16 As[128 * GP];
  __shared__ u16 Bs[128 * GP];
  const int m0 = blockIdx.y * 128, n0 = blockIdx.x * 128;
  const int tid = threadIdx.x, lane = tid & 31, w = tid >> 5;
  const int wy = w >> 2, wx = w & 3, r = lane & 15, hi = lane >> 4;

  v8f acc[4][2] = {};
  gemm128(ohb, wob, m0, n0, As, Bs, acc);

#pragma unroll
  for (int mt = 0; mt < 4; ++mt) {
#pragma unroll
    for (int nt = 0; nt < 2; ++nt) {
      const int n = n0 + wx * 32 + nt * 16 + r;
      const float bias = outb[n];
#pragma unroll
      for (int e = 0; e < 8; ++e) {
        const int m = m0 + wy * 64 + mt * 16 + e + 8 * hi;
        o[(size_t)m * EMB + n] = acc[mt][nt][e] + bias;
      }
    }
  }
}

// ---------------------------------------------------------------------------
// Fused attention: per block = one head x 128 query rows (8 row-tiles).
// K and V^T staged in LDS (CDNA5 320KB WGP LDS). Scores computed transposed
// (S^T = K.Q^T) so softmax stats are per-lane and the score D-frags map 1:1
// onto P@V A-fragments. 3-pass (max / sum+P.V / normalized attn write).
// ---------------------------------------------------------------------------
#define KSH_P 72     // k LDS pitch (64+8)
#define VT_P  1032   // v^T LDS pitch (1024+8)
#define SMEM_ATTN (SEQ * KSH_P * 2 + DH * VT_P * 2 + (128 + 128 + 32) * 4 + 8 * 1024 * 4)

#if defined(HAS_TDM)
// TDM copy of the full 1024x64 bf16 K tile into LDS; the TDM pad feature
// reproduces the 72-element LDS pitch: one row = 128B = 32 DWORDs
// (pad_interval code 4), pad = 16B = 4 DWORDs (pad_amount code 3).
DEV void tdm_load_k(const u16* kg, u16* ksh) {
  u32x4 g0 = {};
  i32x8 g1 = {};
  i32x4 g2 = {};
  i32x4 g3 = {};
  const u64 ga = (u64)(size_t)kg;
  g0[0] = 1u;                                             // count=1, user D#
  g0[1] = (u32)(size_t)ksh;                               // lds_addr (bytes)
  g0[2] = (u32)ga;                                        // global_addr[31:0]
  g0[3] = ((u32)(ga >> 32) & 0x01ffffffu) | (2u << 30);   // addr[56:32]|type=2
  g1[0] = (int)((1u << 16) |                              // data_size = 2B
                (1u << 20) |                              // pad_enable
                (4u << 22) |                              // pad_interval: 32 dw
                (3u << 25));                              // pad_amount: 4 dw
  g1[1] = (int)(64u << 16);                               // tensor_dim0 = 64
  g1[2] = (int)(1024u << 16);                             // tensor_dim1 = 1024
  g1[3] = (int)(64u << 16);                               // tile_dim0 = 64
  g1[4] = 1024;                                           // tile_dim1 = 1024
  g1[5] = 64;                                             // dim0_stride = 64
  g1[6] = 0;
  g1[7] = 0;
#if TDM_ARITY == 6
  i32x8 g4 = {};
  __builtin_amdgcn_tensor_load_to_lds(g0, g1, g2, g3, g4, 0);
#else
  __builtin_amdgcn_tensor_load_to_lds(g0, g1, g2, g3, 0);
#endif
}
#endif

DEV v8f score16(const u16* ksh, int j0, v16bf q0, v16bf q1, int lane) {
  v16bf ka = load_frag(ksh, KSH_P, j0, 0, lane);
  v16bf kc = load_frag(ksh, KSH_P, j0, 32, lane);
  v8f d = {};
  d = wmma_bf(ka, q0, d);
  d = wmma_bf(kc, q1, d);
  return d;
}

__global__ __launch_bounds__(256) void attn_kernel(
    const u16* __restrict__ qb, const u16* __restrict__ kb,
    const u16* __restrict__ vb, const int* __restrict__ mask,
    float* __restrict__ attn, u16* __restrict__ ohb) {
  extern __shared__ char smem[];
  u16*   ksh  = (u16*)smem;                                   // [1024][72]
  u16*   vT   = (u16*)(smem + SEQ * KSH_P * 2);               // [64][1032]
  float* wmax = (float*)(smem + SEQ * KSH_P * 2 + DH * VT_P * 2);  // [8][16]
  float* wsum = wmax + 128;                                   // [8][16]
  float* gst  = wsum + 128;    // [0..15]=row max, [16..31]=1/row sum
  float* opart = gst + 32;     // [8][16][64] per-wave o partials

  const int g = blockIdx.y;            // head id = b*NH + h
  const int bb = g >> 4, hh = g & 15;
  const int tid = threadIdx.x, lane = tid & 31, w = tid >> 5;
  const int r = lane & 15, hi = lane >> 4;
  const int jbase = w * 128;           // this wave's attn-column range

  const u16* kg = kb + (size_t)g * (SEQ * DH);
  const u16* vg = vb + (size_t)g * (SEQ * DH);
  const u16* qg = qb + (size_t)g * (SEQ * DH);
  const int* mrow = mask + bb * SEQ;

  // Stage K (row-major) into LDS: TDM > async > manual
#if defined(HAS_TDM)
  if (tid < 32) tdm_load_k(kg, ksh);
#elif defined(HAS_ASYNC)
  for (int t = tid; t < 8192; t += 256) {
    const int row = t >> 3, kq = (t & 7) * 8;
    async_copy16(ksh + row * KSH_P + kq, kg + (size_t)row * DH + kq);
  }
#else
  for (int t = tid; t < 8192; t += 256) {
    const int row = t >> 3, kq = (t & 7) * 8;
    *(i4x*)(ksh + row * KSH_P + kq) = *(const i4x*)(kg + (size_t)row * DH + kq);
  }
#endif
  // Stage V transposed: vT[d][j] = v[j][d]
  for (int t = tid; t < 32768; t += 256) {
    const int e = t << 1;
    const int j = e >> 6, d = e & 63;
    const u32 val = *(const u32*)(vg + (size_t)j * DH + d);
    vT[(d) * VT_P + j]     = (u16)(val & 0xffffu);
    vT[(d + 1) * VT_P + j] = (u16)(val >> 16);
  }
#if defined(HAS_TDM)
  if (tid < 32) __builtin_amdgcn_s_wait_tensorcnt(0);
#elif defined(HAS_ASYNC)
  wait_async0();
#endif
  __syncthreads();

  for (int it = 0; it < 8; ++it) {
    const int i0 = blockIdx.x * 128 + it * 16;

    // Q fragments (B-matrix layout, lane = query row), kept in registers
    const v16bf q0 = load_frag(qg, DH, i0, 0, lane);
    const v16bf q1 = load_frag(qg, DH, i0, 32, lane);
    const int mi = mrow[i0 + r];

    // ---- pass 1: row max ----
    float rm = -3.0e38f;
    for (int jt = 0; jt < 128; jt += 16) {
      const int j0 = jbase + jt;
      const v8f d0 = score16(ksh, j0, q0, q1, lane);
      const int* pj = mrow + j0 + hi * 8;
#pragma unroll
      for (int e = 0; e < 8; ++e) {
        const float s = d0[e] + ((mi != 0 && pj[e] != 0) ? 0.f : NEGBIG);
        rm = fmaxf(rm, s);
      }
    }
    rm = fmaxf(rm, __shfl_xor(rm, 16, 32));
    if (hi == 0) wmax[w * 16 + r] = rm;
    __syncthreads();
    if (tid < 16) {
      float mm = wmax[tid];
#pragma unroll
      for (int ww = 1; ww < 8; ++ww) mm = fmaxf(mm, wmax[ww * 16 + tid]);
      gst[tid] = mm;
    }
    __syncthreads();
    const float gmax = gst[r];

    // ---- pass 2: exp-sum and unnormalized P@V ----
    float rs = 0.f;
    v8f ov[4] = {};
    for (int jc = 0; jc < 128; jc += 32) {
      const int j0 = jbase + jc;
      const v8f d0 = score16(ksh, j0, q0, q1, lane);
      const v8f d1 = score16(ksh, j0 + 16, q0, q1, lane);
      float p0[8], p1[8];
      const int* pj0 = mrow + j0 + hi * 8;
      const int* pj1 = mrow + j0 + 16 + hi * 8;
#pragma unroll
      for (int e = 0; e < 8; ++e) {
        float s = d0[e] + ((mi != 0 && pj0[e] != 0) ? 0.f : NEGBIG);
        p0[e] = __expf(s - gmax);
        rs += p0[e];
        s = d1[e] + ((mi != 0 && pj1[e] != 0) ? 0.f : NEGBIG);
        p1[e] = __expf(s - gmax);
        rs += p1[e];
      }
      // Score^T D-frags map directly onto the P A-fragment layout.
      FragU ap;
#pragma unroll
      for (int e = 0; e < 8; ++e) {
        ap.u[e] = f2bf(p0[e]);
        ap.u[8 + e] = f2bf(p1[e]);
      }
#pragma unroll
      for (int nd = 0; nd < 4; ++nd) {
        const v16bf bv = load_frag(vT, VT_P, nd * 16, j0, lane);
        ov[nd] = wmma_bf(ap.v, bv, ov[nd]);
      }
    }
    rs += __shfl_xor(rs, 16, 32);
    if (hi == 0) wsum[w * 16 + r] = rs;
    // Deterministic cross-wave o reduction: per-wave slots, no float atomics
#pragma unroll
    for (int nd = 0; nd < 4; ++nd)
#pragma unroll
      for (int e = 0; e < 8; ++e)
        opart[w * 1024 + (e + 8 * hi) * 64 + nd * 16 + r] = ov[nd][e];
    __syncthreads();
    if (tid < 16) {
      float ss = 0.f;
#pragma unroll
      for (int ww = 0; ww < 8; ++ww) ss += wsum[ww * 16 + tid];
      gst[16 + tid] = 1.0f / ss;
    }
    __syncthreads();

    // o-head tile (bf16, normalized), layout [(s*B+b)][h*64+d]
    for (int t = tid; t < 1024; t += 256) {
      const int ir = t >> 6, d = t & 63;
      float val = 0.f;
#pragma unroll
      for (int ww = 0; ww < 8; ++ww) val += opart[ww * 1024 + t];
      val *= gst[16 + ir];
      ohb[((size_t)(i0 + ir) * BATCH + bb) * EMB + hh * DH + d] = f2bf(val);
    }

    // ---- pass 3: write normalized attention (two b128 stores per lane/tile)
    const float invl = gst[16 + r];
    for (int jt = 0; jt < 128; jt += 16) {
      const int j0 = jbase + jt;
      const v8f d0 = score16(ksh, j0, q0, q1, lane);
      const int* pj = mrow + j0 + hi * 8;
      f4x lo, hv;
#pragma unroll
      for (int e = 0; e < 8; ++e) {
        const float s = d0[e] + ((mi != 0 && pj[e] != 0) ? 0.f : NEGBIG);
        const float p = __expf(s - gmax) * invl;
        if (e < 4) lo[e] = p; else hv[e - 4] = p;
      }
      const size_t idx = ((size_t)g * SEQ + (i0 + r)) * SEQ + j0 + hi * 8;
      *(f4x*)(attn + idx) = lo;
      *(f4x*)(attn + idx + 4) = hv;
    }
    __syncthreads();
  }
}

// ---------------------------------------------------------------------------
// Host launcher
// ---------------------------------------------------------------------------
extern "C" void kernel_launch(void* const* d_in, const int* in_sizes, int n_in,
                              void* d_out, int out_size, void* d_ws, size_t ws_size,
                              hipStream_t stream) {
  const float* x     = (const float*)d_in[0];
  const int*   mask  = (const int*)d_in[1];
  const float* qkv_w = (const float*)d_in[2];
  const float* qkv_b = (const float*)d_in[3];
  const float* out_w = (const float*)d_in[4];
  const float* out_b = (const float*)d_in[5];
  float* out = (float*)d_out;          // [0, 8.4M) = o ; [8.4M, ...) = attn

  char* ws = (char*)d_ws;
  u16* xb  = (u16*)(ws);                                  // 16 MB
  u16* wqb = (u16*)(ws + 16777216);                       // 6  MB
  u16* wob = (u16*)(ws + 16777216 + 6291456);             // 2  MB
  u16* qb  = (u16*)(ws + 25165824);                       // 16 MB
  u16* kb  = (u16*)(ws + 25165824 + 16777216);            // 16 MB
  u16* vb  = (u16*)(ws + 25165824 + 2 * 16777216);        // 16 MB
  u16* ohb = (u16*)(ws + 25165824 + 3 * 16777216);        // 16 MB

  cvt_bf16_kernel<<<8388608 / 256, 256, 0, stream>>>(x, xb, 8388608);
  cvt_bf16_kernel<<<3145728 / 256, 256, 0, stream>>>(qkv_w, wqb, 3145728);
  cvt_bf16_kernel<<<1048576 / 256, 256, 0, stream>>>(out_w, wob, 1048576);

  gemm_qkv_kernel<<<dim3(24, 64), 256, 0, stream>>>(xb, wqb, qkv_b, mask,
                                                    qb, kb, vb);

  (void)hipFuncSetAttribute((const void*)attn_kernel,
                            hipFuncAttributeMaxDynamicSharedMemorySize,
                            SMEM_ATTN);
  attn_kernel<<<dim3(8, 128), 256, SMEM_ATTN, stream>>>(
      qb, kb, vb, mask, out + (size_t)SEQ * BATCH * EMB, ohb);

  gemm_out_kernel<<<dim3(8, 64), 256, 0, stream>>>(ohb, wob, out_b, out);
}
